// SelfAttention_88802743812437
// MI455X (gfx1250) — matrix operand order repaired and checked by reading, not compiled, not run on hardware
//
#include <hip/hip_runtime.h>
#include <math.h>

typedef __attribute__((ext_vector_type(16))) _Float16 v16h;
typedef __attribute__((ext_vector_type(8)))  float    v8f;
typedef __attribute__((ext_vector_type(4)))  float    v4f;

#define SEQ 2048
#define DIM 64
#define QT  16          // q rows per block
#define NWAVE 8
#define NTHREADS 256

// LDS layout (floats):
//   sc      : 16*2048  score/prob slab (16B-aligned rows, stride 2048)
//   red     : 16*16    per-row partial reductions
//   rowmax  : 16
//   rowinv  : 16
//   part    : 2*16*64  PV partial accumulators (two k-halves)
#define LDS_FLOATS (QT*SEQ + 256 + 16 + 16 + 2*QT*DIM)

__global__ __launch_bounds__(NTHREADS) void
attn_fwd_wmma_kernel(const float* __restrict__ Kmat,
                     const float* __restrict__ Qmat,
                     const float* __restrict__ Vmat,
                     const int*   __restrict__ mask,
                     float* __restrict__ ctx_out,
                     float* __restrict__ attn_out)
{
    extern __shared__ float lds[];
    float* sc     = lds;                   // [16][2048]
    float* red    = sc + QT*SEQ;           // [16][16]
    float* rowmax = red + 256;             // [16]
    float* rowinv = rowmax + 16;           // [16]
    float* part   = rowinv + 16;           // [2][16][64]

    const int tid  = threadIdx.x;
    const int wave = tid >> 5;
    const int lane = tid & 31;
    const int b    = blockIdx.x >> 7;      // 128 q-tiles per batch
    const int qt   = blockIdx.x & 127;

    const int lrow = lane & 15;            // A row / B column / C column index
    const int lhi  = lane >> 4;            // half-wave selector

    // ---------------- Q fragments (A-matrix, 16x32 f16 layout, 2 K-chunks) ----
    const float* qrow = Qmat + ((size_t)(b*SEQ + qt*QT + lrow)) * DIM;
    v16h aq[2];
#pragma unroll
    for (int c = 0; c < 2; ++c) {
        const float* p0 = qrow + c*32 + lhi*16;
#pragma unroll
        for (int j = 0; j < 8; ++j) {
            aq[c][j]     = (_Float16)p0[j];
            aq[c][8 + j] = (_Float16)p0[8 + j];
        }
    }

    // ---------------- scores = Q K^T / 8, masked, into LDS ---------------------
    const float* kbatch = Kmat + (size_t)b * SEQ * DIM;
    const int*   mrow   = mask + b * SEQ;
    for (int kt = wave; kt < SEQ/16; kt += NWAVE) {
        const float* kptr = kbatch + (size_t)(kt*16 + lrow) * DIM;
        v8f acc = {};
#pragma unroll
        for (int c = 0; c < 2; ++c) {
            const float* p = kptr + c*32 + lhi*16;   // B: 16 contiguous K values
            v16h bk;
#pragma unroll
            for (int j = 0; j < 16; ++j) bk[j] = (_Float16)p[j];
            acc = __builtin_amdgcn_wmma_f32_16x16x32_f16(
                      false, aq[c], false, bk, (short)0, acc, false, false);
        }
        const int  kcol   = kt*16 + lrow;            // this lane's key index
        const bool masked = (mrow[kcol] != 0);       // mask==True -> -inf
#pragma unroll
        for (int i = 0; i < 8; ++i) {
            float s = acc[i] * 0.125f;               // 1/sqrt(64)
            if (masked) s = -INFINITY;
            sc[(i + lhi*8) * SEQ + kcol] = s;        // C layout: M = i + lhi*8
        }
    }
    __syncthreads();

    // ---------------- softmax over k, in-place in LDS, b128 granularity -------
    {
        const int row = tid >> 4;
        const int ln  = tid & 15;
        v4f* scrow = (v4f*)(sc + row*SEQ);           // 512 vectors per row

        float mx = -INFINITY;
        for (int c4 = ln; c4 < SEQ/4; c4 += 16) {
            v4f v = scrow[c4];
            mx = fmaxf(mx, fmaxf(fmaxf(v[0], v[1]), fmaxf(v[2], v[3])));
        }
        red[row*16 + ln] = mx;
        __syncthreads();
        if (ln == 0) {
            float m = red[row*16];
#pragma unroll
            for (int j = 1; j < 16; ++j) m = fmaxf(m, red[row*16 + j]);
            if (m == -INFINITY) m = 0.0f;            // all-masked guard
            rowmax[row] = m;
        }
        __syncthreads();
        const float rm = rowmax[row];
        float s = 0.0f;
        for (int c4 = ln; c4 < SEQ/4; c4 += 16) {
            v4f v = scrow[c4];
            v4f e;
#pragma unroll
            for (int j = 0; j < 4; ++j) e[j] = __expf(v[j] - rm);
            scrow[c4] = e;
            s += (e[0] + e[1]) + (e[2] + e[3]);
        }
        red[row*16 + ln] = s;
        __syncthreads();
        if (ln == 0) {
            float t = 0.0f;
#pragma unroll
            for (int j = 0; j < 16; ++j) t += red[row*16 + j];
            rowinv[row] = (t > 0.0f) ? (1.0f / t) : 0.0f;
        }
        __syncthreads();
    }

    // ------- normalize + stream attention matrix out (b128 NT stores) ---------
    {
        v4f* scv = (v4f*)sc;
        v4f* abase = (v4f*)(attn_out + ((size_t)(b*SEQ) + qt*QT) * SEQ);
        for (int i4 = tid; i4 < QT*SEQ/4; i4 += NTHREADS) {
            const int r = i4 >> 9;                   // row = (i4*4) / 2048
            const float inv = rowinv[r];
            v4f p = scv[i4];
            p *= inv;
            scv[i4] = p;                             // keep normalized P for PV
            __builtin_nontemporal_store(p, abase + i4);
        }
    }
    __syncthreads();

    // ---------------- context = P V --------------------------------------------
    {
        const int nt = wave & 3;                     // d-tile (16 cols of 64)
        const int kh = wave >> 2;                    // k-half
        const float* vbatch = Vmat + (size_t)b * SEQ * DIM;
        const int dcol = nt*16 + lrow;
        v8f acc = {};
        for (int ch = 0; ch < 32; ++ch) {
            const int kb = kh*1024 + ch*32;
            // A fragment: P rows from LDS (16B-aligned vector reads)
            v16h ap;
            const v4f* sp = (const v4f*)(sc + lrow*SEQ + kb + lhi*16);
            v4f a0 = sp[0], a1 = sp[1], a2 = sp[2], a3 = sp[3];
#pragma unroll
            for (int j = 0; j < 4; ++j) {
                ap[j]      = (_Float16)a0[j];
                ap[4 + j]  = (_Float16)a1[j];
                ap[8 + j]  = (_Float16)a2[j];
                ap[12 + j] = (_Float16)a3[j];
            }
            // B fragment: V column dcol, 16 contiguous k (64B-coalesced gather)
            v16h bv;
            const float* vp = vbatch + (size_t)(kb + lhi*16) * DIM + dcol;
#pragma unroll
            for (int j = 0; j < 16; ++j) bv[j] = (_Float16)vp[(size_t)j * DIM];
            acc = __builtin_amdgcn_wmma_f32_16x16x32_f16(
                      false, ap, false, bv, (short)0, acc, false, false);
        }
#pragma unroll
        for (int i = 0; i < 8; ++i)
            part[kh*1024 + (i + lhi*8)*DIM + dcol] = acc[i];
    }
    __syncthreads();

    // combine the two k-halves and write context (one b128 per thread)
    {
        v4f* cbase = (v4f*)(ctx_out + ((size_t)(b*SEQ) + qt*QT) * DIM);
        v4f* p0 = (v4f*)part;
        v4f* p1 = (v4f*)(part + QT*DIM);
        for (int i4 = tid; i4 < QT*DIM/4; i4 += NTHREADS)
            cbase[i4] = p0[i4] + p1[i4];
    }
}

extern "C" void kernel_launch(void* const* d_in, const int* in_sizes, int n_in,
                              void* d_out, int out_size, void* d_ws, size_t ws_size,
                              hipStream_t stream) {
    // setup_inputs order: key, query, value, query_attention_mask
    const float* Kp = (const float*)d_in[0];
    const float* Qp = (const float*)d_in[1];
    const float* Vp = (const float*)d_in[2];
    const int*   Mp = (const int*)  d_in[3];

    float* ctx  = (float*)d_out;                       // [8,2048,64]
    float* attn = ctx + (size_t)8 * SEQ * DIM;         // [8,2048,2048]

    const size_t shmem = (size_t)LDS_FLOATS * sizeof(float);  // ~137 KB
    hipFuncSetAttribute(reinterpret_cast<const void*>(attn_fwd_wmma_kernel),
                        hipFuncAttributeMaxDynamicSharedMemorySize, (int)shmem);

    dim3 grid(8 * (SEQ / QT));   // 1024 blocks: (batch, q-tile)
    dim3 block(NTHREADS);        // 8 waves of 32
    attn_fwd_wmma_kernel<<<grid, block, shmem, stream>>>(Kp, Qp, Vp, Mp, ctx, attn);
}
